// BiLSTM_45810121179417
// MI455X (gfx1250) — matrix operand order repaired and checked by reading, not compiled, not run on hardware
//
#include <hip/hip_runtime.h>
#include <hip/hip_bf16.h>
#include <math.h>

typedef __attribute__((ext_vector_type(16))) _Float16 v16h;
typedef __attribute__((ext_vector_type(8)))  _Float16 v8h;
typedef __attribute__((ext_vector_type(8)))  float    v8f;

#define HID   1024
#define GATES 4096
#define BATCH 16
#define TSEQ  128
#define FIN   1200
#define FINP  1216
#define NLAY  6
#define TPAD  40     // padded LDS row stride (halfs) for 32-wide k slabs: 20 words -> conflict-free
#define HPAD  1032   // padded LDS row stride (halfs) for h-state: 516 words -> conflict-free

// ---------------------------------------------------------------------------
// CDNA5 async global->LDS copy (ASYNCcnt path, CDNA5 ISA).
// Each lane copies 16B from its global address to its LDS address; the LDS
// byte offset is the low 32 bits of the generic shared-memory pointer.
// ---------------------------------------------------------------------------
__device__ __forceinline__ void async_copy_b128(const _Float16* __restrict__ g,
                                                _Float16* __restrict__ l) {
  unsigned lofs = (unsigned)(unsigned long long)l;
  asm volatile("global_load_async_to_lds_b128 %0, %1, off"
               :: "v"(lofs), "v"(g) : "memory");
}
__device__ __forceinline__ void wait_async_le3() {
  asm volatile("s_wait_asynccnt 0x3" ::: "memory");
}
__device__ __forceinline__ void wait_async_0() {
  asm volatile("s_wait_asynccnt 0x0" ::: "memory");
}

// ---------------------------------------------------------------------------
// WMMA tile loaders, faithful to CDNA5 ISA 16-bit layouts (CDNA5 ISA)
// A (16x32, MxK): lane<16 -> M=lane, K in {0..7,16..23}; lane>=16 -> K in {8..15,24..31}
// B (32x16, KxN): lane&15 = N, lane<16 -> K in {0..7,16..23}; lane>=16 -> K in {8..15,24..31}
// C/D (16x16):    lane&15 = N, (lane>>4)*8 + vgpr = M
// ---------------------------------------------------------------------------
__device__ __forceinline__ v16h load_tileA(const _Float16* __restrict__ p0, int ldk) {
  const int lane = threadIdx.x & 31;
  const int r    = lane & 15;
  const int hs   = (lane >> 4) << 3;
  const _Float16* p = p0 + (size_t)r * ldk + hs;
  v8h lo = *(const v8h*)(p);
  v8h hi = *(const v8h*)(p + 16);
  v16h out;
#pragma unroll
  for (int i = 0; i < 8; ++i) { out[i] = lo[i]; out[i + 8] = hi[i]; }
  return out;
}

__device__ __forceinline__ v16h load_tileB(const _Float16* __restrict__ p0, int ldk) {
  const int lane = threadIdx.x & 31;
  const int n    = lane & 15;
  const int kb   = (lane >> 4) << 3;
  const _Float16* p = p0 + (size_t)n * ldk + kb;
  v8h lo = *(const v8h*)(p);
  v8h hi = *(const v8h*)(p + 16);
  v16h out;
#pragma unroll
  for (int i = 0; i < 8; ++i) { out[i] = lo[i]; out[i + 8] = hi[i]; }
  return out;
}

// ---------------------------------------------------------------------------
// Conversion / init kernels
// ---------------------------------------------------------------------------
__global__ void cvt_f32_f16(const float* __restrict__ src, _Float16* __restrict__ dst, size_t n) {
  size_t i = (size_t)blockIdx.x * 256 + threadIdx.x;
  if (i < n) dst[i] = (_Float16)src[i];
}

__global__ void cvt_pad_rows(const float* __restrict__ src, _Float16* __restrict__ dst,
                             int kin, int kout) {
  int r = blockIdx.x;
  for (int k = threadIdx.x; k < kout; k += 256)
    dst[(size_t)r * kout + k] = (k < kin) ? (_Float16)src[(size_t)r * kin + k] : (_Float16)0.f;
}

// X [B,T,F] f32 -> x0p [t*B+b, FINP] f16 (time-major rows for the GEMM)
__global__ void cvt_x0(const float* __restrict__ X, _Float16* __restrict__ dst) {
  int row = blockIdx.x;
  int t = row >> 4, b = row & 15;
  const float* s = X + ((size_t)b * TSEQ + t) * FIN;
  for (int k = threadIdx.x; k < FINP; k += 256)
    dst[(size_t)row * FINP + k] = (k < FIN) ? (_Float16)s[k] : (_Float16)0.f;
}

__global__ void init_state(const float* __restrict__ h0, const float* __restrict__ c0,
                           int layer, _Float16* __restrict__ hbuf0, float* __restrict__ cst) {
  int i = blockIdx.x * 256 + threadIdx.x;
  if (i >= 2 * BATCH * HID) return;
  int d = i / (BATCH * HID);
  size_t src = ((size_t)(2 * layer + d)) * BATCH * HID + (size_t)(i % (BATCH * HID));
  hbuf0[i] = (_Float16)h0[src];
  cst[i]   = c0[src];
}

// ---------------------------------------------------------------------------
// Input projection GEMM:  Y[dir][m][n] = bias[dir][n] + sum_k X[m][k]*W[dir][n][k]
// Block = 8 waves on a 128(M) x 64(N) tile; wave = 32x32 (2x2 WMMA tiles).
// K staged through double-buffered LDS via async global->LDS copies:
//   per k-chunk (32), A slab 128x32 (2 async-b128/thread) + B slab 64x32 (1/thread)
//   -> exactly 3 ASYNCcnt per wave per stage; overlap next stage with WMMA.
// grid = (N/64, M/128, 2 dirs)
// ---------------------------------------------------------------------------
__global__ __launch_bounds__(256)
void proj_gemm(const _Float16* __restrict__ X, const _Float16* __restrict__ W,
               const float* __restrict__ bias, float* __restrict__ Y, int K) {
  __shared__ _Float16 As[2][128 * TPAD];
  __shared__ _Float16 Bs[2][64 * TPAD];

  const int dir = blockIdx.z;
  const _Float16* Wd = W + (size_t)dir * GATES * K;
  const float*    bd = bias + (size_t)dir * GATES;
  float*          Yd = Y + (size_t)dir * (TSEQ * BATCH) * GATES;

  const int tid  = threadIdx.x;
  const int w    = tid >> 5;
  const int lane = tid & 31;
  const int mB = blockIdx.y * 128;
  const int nB = blockIdx.x * 64;
  const int m0l = (w & 3) * 32;      // wave tile origin within block
  const int n0l = (w >> 2) * 32;

  v8f acc[2][2];
#pragma unroll
  for (int mi = 0; mi < 2; ++mi)
#pragma unroll
    for (int ni = 0; ni < 2; ++ni) {
      float bv = bd[nB + n0l + ni * 16 + (lane & 15)];
#pragma unroll
      for (int v = 0; v < 8; ++v) acc[mi][ni][v] = bv;
    }

  // async stage of one 32-wide k slab into buffer `buf`
  const int ar0 = tid >> 2;                 // A: 4 x 16B chunks per 32-half row
  const int ac0 = (tid & 3) * 8;
  auto stage = [&](int buf, int k) {
#pragma unroll
    for (int i = 0; i < 2; ++i) {
      int row = ar0 + i * 64;
      async_copy_b128(X + (size_t)(mB + row) * K + k + ac0, &As[buf][row * TPAD + ac0]);
    }
    async_copy_b128(Wd + (size_t)(nB + ar0) * K + k + ac0, &Bs[buf][ar0 * TPAD + ac0]);
    if (k + 256 < K) {                      // warm GL2 further ahead
      __builtin_prefetch(X  + (size_t)(mB + ar0) * K + k + 256, 0, 0);
      __builtin_prefetch(Wd + (size_t)(nB + ar0) * K + k + 256, 0, 0);
    }
  };

  stage(0, 0);
  int buf = 0;
  for (int k = 0; k < K; k += 32) {
    if (k + 32 < K) { stage(buf ^ 1, k + 32); wait_async_le3(); }
    else            { wait_async_0(); }
    __syncthreads();

    v16h a0 = load_tileA(&As[buf][(m0l)      * TPAD], TPAD);
    v16h a1 = load_tileA(&As[buf][(m0l + 16) * TPAD], TPAD);
    v16h b0 = load_tileB(&Bs[buf][(n0l)      * TPAD], TPAD);
    v16h b1 = load_tileB(&Bs[buf][(n0l + 16) * TPAD], TPAD);
    acc[0][0] = __builtin_amdgcn_wmma_f32_16x16x32_f16(false, a0, false, b0, (short)0, acc[0][0], false, false);
    acc[0][1] = __builtin_amdgcn_wmma_f32_16x16x32_f16(false, a0, false, b1, (short)0, acc[0][1], false, false);
    acc[1][0] = __builtin_amdgcn_wmma_f32_16x16x32_f16(false, a1, false, b0, (short)0, acc[1][0], false, false);
    acc[1][1] = __builtin_amdgcn_wmma_f32_16x16x32_f16(false, a1, false, b1, (short)0, acc[1][1], false, false);

    __syncthreads();
    buf ^= 1;
  }

  const int nloc  = lane & 15;
  const int mbase = (lane >> 4) * 8;
#pragma unroll
  for (int mi = 0; mi < 2; ++mi)
#pragma unroll
    for (int ni = 0; ni < 2; ++ni)
#pragma unroll
      for (int v = 0; v < 8; ++v) {
        int m = mB + m0l + mi * 16 + mbase + v;
        int n = nB + n0l + ni * 16 + nloc;
        Yd[(size_t)m * GATES + n] = acc[mi][ni][v];
      }
}

// ---------------------------------------------------------------------------
// Fused LSTM step (both directions in one launch):
//   gates = x_proj[t] + h_in @ Whh^T ;  c,h update ;  h -> h_out and x_next
// grid = 32 blocks (dir x 16 column-blocks of 64 h-units), 8 waves each.
// h_in (16x1024 f16, shared by all waves) is async-staged into LDS once.
// ---------------------------------------------------------------------------
__global__ __launch_bounds__(256)
void lstm_step(const float* __restrict__ xproj, const _Float16* __restrict__ Whh,
               const _Float16* __restrict__ h_in, _Float16* __restrict__ h_out,
               float* __restrict__ c_state, _Float16* __restrict__ x_next, int s) {
  __shared__ _Float16 hA[BATCH * HPAD];         // 33 KB staged h-state
  __shared__ float gl[4][16][64];               // 16 KB gate staging
  const int dir = blockIdx.x >> 4;
  const int cb  = (blockIdx.x & 15) * 64;
  const int t   = dir ? (TSEQ - 1 - s) : s;
  const int w    = threadIdx.x >> 5;
  const int lane = threadIdx.x & 31;
  const int nloc  = lane & 15;
  const int mbase = (lane >> 4) * 8;

  const _Float16* hs_ = h_in + (size_t)dir * BATCH * HID;
  const _Float16* Wd  = Whh  + (size_t)dir * GATES * HID;
  const float*    xp  = xproj + ((size_t)dir * TSEQ * BATCH + (size_t)t * BATCH) * GATES;

  // async-stage h (16 x 1024 halfs = 2048 x 16B) into LDS
#pragma unroll
  for (int i = 0; i < 8; ++i) {
    int c   = threadIdx.x + i * 256;            // 0..2047
    int row = c >> 7;                           // 128 x 16B per row
    int c8  = (c & 127) * 8;
    async_copy_b128(hs_ + (size_t)row * HID + c8, &hA[row * HPAD + c8]);
  }

  // two 16x16 gate tiles per wave (16 tiles = 4 gates x 4 sub-tiles of 16 cols)
  int gate[2], sub[2], n0[2];
#pragma unroll
  for (int tt = 0; tt < 2; ++tt) {
    int tile = w * 2 + tt;
    gate[tt] = tile >> 2;
    sub[tt]  = tile & 3;
    n0[tt]   = gate[tt] * HID + cb + sub[tt] * 16;
  }

  v8f acc[2];
#pragma unroll
  for (int tt = 0; tt < 2; ++tt)
#pragma unroll
    for (int v = 0; v < 8; ++v)
      acc[tt][v] = xp[(size_t)(mbase + v) * GATES + n0[tt] + nloc];

  wait_async_0();
  __syncthreads();

  for (int k = 0; k < HID; k += 32) {
    v16h a  = load_tileA(&hA[k], HPAD);                        // shared A from LDS
    v16h b0 = load_tileB(Wd + (size_t)n0[0] * HID + k, HID);
    v16h b1 = load_tileB(Wd + (size_t)n0[1] * HID + k, HID);
    acc[0] = __builtin_amdgcn_wmma_f32_16x16x32_f16(false, a, false, b0, (short)0, acc[0], false, false);
    acc[1] = __builtin_amdgcn_wmma_f32_16x16x32_f16(false, a, false, b1, (short)0, acc[1], false, false);
  }

#pragma unroll
  for (int tt = 0; tt < 2; ++tt)
#pragma unroll
    for (int v = 0; v < 8; ++v)
      gl[gate[tt]][mbase + v][sub[tt] * 16 + nloc] = acc[tt][v];

  __syncthreads();

  // fused LSTM cell: 16 rows x 64 cols, 4 per thread
#pragma unroll
  for (int r = 0; r < 4; ++r) {
    int e   = threadIdx.x + 256 * r;
    int m   = e >> 6;
    int col = e & 63;
    float iv = gl[0][m][col], fv = gl[1][m][col], gv = gl[2][m][col], ov = gl[3][m][col];
    float ig = 1.f / (1.f + __expf(-iv));
    float fg = 1.f / (1.f + __expf(-fv));
    float gg = tanhf(gv);
    float og = 1.f / (1.f + __expf(-ov));
    size_t ci = (size_t)dir * BATCH * HID + (size_t)m * HID + cb + col;
    float cnew = fg * c_state[ci] + ig * gg;
    c_state[ci] = cnew;
    float hnew = og * tanhf(cnew);
    h_out[ci] = (_Float16)hnew;
    x_next[((size_t)t * BATCH + m) * (2 * HID) + (size_t)dir * HID + cb + col] = (_Float16)hnew;
  }
}

// ---------------------------------------------------------------------------
// Tiny MLP head
// ---------------------------------------------------------------------------
__global__ void fc1_kernel(const _Float16* __restrict__ xlast, const float* __restrict__ w,
                           const float* __restrict__ b, float* __restrict__ out1) {
  int br = blockIdx.x;
  int j  = threadIdx.x;
  const _Float16* xr = xlast + (size_t)((TSEQ - 1) * BATCH + br) * (2 * HID);
  const float*    wr = w + (size_t)j * (2 * HID);
  float s = b[j];
  for (int k = 0; k < 2 * HID; ++k) s += (float)xr[k] * wr[k];
  out1[br * 256 + j] = s;
}

__global__ void fc2_kernel(const float* __restrict__ out1, const float* __restrict__ w2,
                           const float* __restrict__ b2, float* __restrict__ out) {
  int br = threadIdx.x;
  if (br >= BATCH) return;
  float s = b2[0];
  for (int k = 0; k < 256; ++k) s += out1[br * 256 + k] * w2[k];
  float t = tanhf(s);
  out[br] = 1.f / (1.f + __expf(-t));
}

// ---------------------------------------------------------------------------
// Host orchestration
// ---------------------------------------------------------------------------
extern "C" void kernel_launch(void* const* d_in, const int* in_sizes, int n_in,
                              void* d_out, int out_size, void* d_ws, size_t ws_size,
                              hipStream_t stream) {
  const float* X        = (const float*)d_in[0];
  const float* h0       = (const float*)d_in[1];
  const float* c0       = (const float*)d_in[2];
  const float* W_ih_l0  = (const float*)d_in[3];
  const float* W_hh_l0  = (const float*)d_in[4];
  const float* b_l0     = (const float*)d_in[5];
  const float* W_ih_r   = (const float*)d_in[6];
  const float* W_hh_r   = (const float*)d_in[7];
  const float* b_rest   = (const float*)d_in[8];
  const float* fc1_w    = (const float*)d_in[9];
  const float* fc1_b    = (const float*)d_in[10];
  const float* fc2_w    = (const float*)d_in[11];
  const float* fc2_b    = (const float*)d_in[12];
  float* out = (float*)d_out;

  char* ws = (char*)d_ws;
  size_t off = 0;
  auto take = [&](size_t bytes) { char* p = ws + off; off += (bytes + 255) & ~(size_t)255; return p; };

  _Float16* wih0p = (_Float16*)take((size_t)2  * GATES * FINP   * 2);
  _Float16* wihR  = (_Float16*)take((size_t)10 * GATES * 2*HID  * 2);
  _Float16* whh   = (_Float16*)take((size_t)12 * GATES * HID    * 2);
  _Float16* x0p   = (_Float16*)take((size_t)TSEQ*BATCH * FINP   * 2);
  _Float16* xA    = (_Float16*)take((size_t)TSEQ*BATCH * 2*HID  * 2);
  _Float16* xB    = (_Float16*)take((size_t)TSEQ*BATCH * 2*HID  * 2);
  float*    xproj = (float*)   take((size_t)2 * TSEQ*BATCH * GATES * 4);
  _Float16* hbuf  = (_Float16*)take((size_t)2 * 2 * BATCH * HID * 2);
  float*    cst   = (float*)   take((size_t)2 * BATCH * HID * 4);
  float*    fc1o  = (float*)   take((size_t)BATCH * 256 * 4);
  (void)ws_size; (void)n_in; (void)in_sizes; (void)out_size;

  cvt_pad_rows<<<2 * GATES, 256, 0, stream>>>(W_ih_l0, wih0p, FIN, FINP);
  {
    size_t n = (size_t)10 * GATES * 2 * HID;
    cvt_f32_f16<<<(int)((n + 255) / 256), 256, 0, stream>>>(W_ih_r, wihR, n);
  }
  {
    size_t n = (size_t)2 * GATES * HID;
    cvt_f32_f16<<<(int)((n + 255) / 256), 256, 0, stream>>>(W_hh_l0, whh, n);
    size_t n2 = (size_t)10 * GATES * HID;
    cvt_f32_f16<<<(int)((n2 + 255) / 256), 256, 0, stream>>>(W_hh_r, whh + n, n2);
  }
  cvt_x0<<<TSEQ * BATCH, 256, 0, stream>>>(X, x0p);

  const size_t HSZ = (size_t)2 * BATCH * HID;

  for (int layer = 0; layer < NLAY; ++layer) {
    const _Float16* xin;
    _Float16* xout = (layer % 2 == 0) ? xA : xB;
    const _Float16* wih;
    const float* bias;
    int K;
    if (layer == 0) { xin = x0p; wih = wih0p; bias = b_l0; K = FINP; }
    else {
      xin  = (layer % 2 == 1) ? xA : xB;
      wih  = wihR + (size_t)(layer - 1) * 2 * GATES * 2 * HID;
      bias = b_rest + (size_t)(layer - 1) * 2 * GATES;
      K    = 2 * HID;
    }

    proj_gemm<<<dim3(GATES / 64, (TSEQ * BATCH) / 128, 2), 256, 0, stream>>>(xin, wih, bias, xproj, K);

    init_state<<<(2 * BATCH * HID + 255) / 256, 256, 0, stream>>>(h0, c0, layer, hbuf, cst);

    const _Float16* whh_l = whh + (size_t)layer * 2 * GATES * HID;
    for (int s = 0; s < TSEQ; ++s) {
      _Float16* hin  = hbuf + (size_t)(s & 1) * HSZ;
      _Float16* hout = hbuf + (size_t)((s & 1) ^ 1) * HSZ;
      lstm_step<<<32, 256, 0, stream>>>(xproj, whh_l, hin, hout, cst, xout, s);
    }
  }

  fc1_kernel<<<BATCH, 256, 0, stream>>>(xB, fc1_w, fc1_b, fc1o);
  fc2_kernel<<<1, 32, 0, stream>>>(fc1o, fc2_w, fc2_b, out);
}
